// GraphNeuralNetworkV2_40913858462138
// MI455X (gfx1250) — matrix operand order repaired and checked by reading, not compiled, not run on hardware
//
#include <hip/hip_runtime.h>
#include <hip/hip_bf16.h>

typedef __attribute__((ext_vector_type(16))) _Float16 v16h;
typedef __attribute__((ext_vector_type(8)))  _Float16 h8;
typedef __attribute__((ext_vector_type(8)))  float    v8f;

// global (addrspace 1) pointer to a 32B B-fragment: forces global_load_b128 (not flat)
typedef const v16h __attribute__((address_space(1))) * gv16h_ptr;

#define N_VAR  50000
#define N_CONS 50000
#define N_EDGE 800000
#define BN_EPS 1e-5f

// ---------------- WMMA fragment helpers (wave32, 16x16x32 f16) ----------------

// A fragment: 16x32 (MxK) from LDS row-major [16][ldk].
// Lanes 0-15: M=lane, K=k0+0..7 and k0+16..23; lanes 16-31: M=lane-16, K=k0+8..15 / k0+24..31.
__device__ inline v16h load_a_frag(const _Float16* tile, int ldk, int k0) {
  int lane = threadIdx.x & 31;
  const _Float16* p = tile + (lane & 15) * ldk + k0 + ((lane >> 4) << 3);
  v16h a;
#pragma unroll
  for (int i = 0; i < 8; ++i) a[i] = p[i];
#pragma unroll
  for (int i = 0; i < 8; ++i) a[8 + i] = p[16 + i];
  return a;
}

// B fragment from pre-swizzled fragment-major weights: one aligned 32B global load per lane.
__device__ inline v16h load_b_packed(const _Float16* Wp, int frag) {
  return *(gv16h_ptr)(Wp + ((size_t)frag * 32 + (threadIdx.x & 31)) * 16);
}

__device__ inline v8f wmma_f16(v16h a, v16h b, v8f c) {
  return __builtin_amdgcn_wmma_f32_16x16x32_f16(false, a, false, b, (short)0, c, false, false);
}

// ---------------- small utility kernels ----------------

__global__ void k_zero_f32(float* __restrict__ p, int n) {
  int i = blockIdx.x * blockDim.x + threadIdx.x;
  if (i < n) p[i] = 0.f;
}

__global__ void k_f32_to_f16(_Float16* __restrict__ dst, const float* __restrict__ src, int n) {
  int i = blockIdx.x * blockDim.x + threadIdx.x;
  if (i < n) dst[i] = (_Float16)src[i];
}

// Pack f32 weight [Kin x N] into f16 WMMA-B fragment-major layout [Kpad/32][N/16][32][16],
// zero-padding K rows. Optional (vf|cf) row swap for the cons messenger (shared A-tile).
// Optional bias: written as weight row Kin (pairs with a constant-1.0 A column) so the
// bias-add is folded into the GEMM itself.
__global__ void k_pack_w(_Float16* __restrict__ dst, const float* __restrict__ src,
                         const float* __restrict__ bias,
                         int Kin, int Kpad, int N, int swap_dv) {
  int i = blockIdx.x * blockDim.x + threadIdx.x;
  int total = Kpad * N;
  if (i >= total) return;
  int h    = i & 15;
  int lane = (i >> 4) & 31;
  int frag = i >> 9;
  int NT = N >> 4;
  int nt = frag % NT;
  int kc = frag / NT;
  int row = kc * 32 + ((lane >> 4) << 3) + (h & 7) + ((h >> 3) << 4);
  int col = nt * 16 + (lane & 15);
  float v = 0.f;
  if (row < Kin) {
    int sr = row;
    if (swap_dv > 0) {
      if (row < swap_dv)          sr = row + swap_dv;
      else if (row < 2 * swap_dv) sr = row - swap_dv;
    }
    v = src[(size_t)sr * N + col];
  } else if (bias != nullptr && row == Kin) {
    v = bias[col];
  }
  dst[i] = (_Float16)v;
}

// BatchNorm finalize+apply; also writes next-iteration f16 feature table.
__global__ void k_bn_apply(const float* __restrict__ x,
                           const float* __restrict__ sum, const float* __restrict__ ssum,
                           const float* __restrict__ gamma, const float* __restrict__ beta,
                           float* __restrict__ outf, _Float16* __restrict__ outh,
                           int n, float invN) {
  int i = blockIdx.x * blockDim.x + threadIdx.x;
  if (i >= n) return;
  int c = i & 63;
  float mu  = sum[c] * invN;
  float var = ssum[c] * invN - mu * mu;
  float r   = rsqrtf(var + BN_EPS);
  float y   = (x[i] - mu) * r * gamma[c] + beta[c];
  outf[i] = y;
  if (outh) outh[i] = (_Float16)y;
}

// ---------------- fused edge-messenger kernel (both sides) ----------------
// Per 16-edge tile (one wave, private LDS slice): gather [vf|cf|ef|1.0] once (f16, 16B
// chunks), then per side: (16xKPAD)@(KPADx128) [bias folded as K-row] -> relu ->
// (16x128)@(128x64)+b2 -> atomic scatter-add segment sum. A-fragments are register-
// resident; B-fragments are single 32B global loads; accumulators start from inline-0.
template<int DV, int KPAD>
__global__ __launch_bounds__(128)
void k_edge_mlp(const _Float16* __restrict__ vf16, const _Float16* __restrict__ cf16,
                const _Float16* __restrict__ ef16,
                const int* __restrict__ ev, const int* __restrict__ ec,
                const _Float16* __restrict__ W1v,
                const _Float16* __restrict__ W2v, const float* __restrict__ b2v,
                const _Float16* __restrict__ W1c,
                const _Float16* __restrict__ W2c, const float* __restrict__ b2c,
                float* __restrict__ aggv, float* __restrict__ aggc) {
  constexpr int KC1 = KPAD / 32;
  __shared__ _Float16 sA[4][16][KPAD];
  __shared__ _Float16 sH[4][16][128];
  int wave = threadIdx.x >> 5;
  int lane = threadIdx.x & 31;
  int e0 = (blockIdx.x * 4 + wave) * 16;

  _Float16 (*A)[KPAD] = sA[wave];
  _Float16 (*H)[128]  = sH[wave];

  { // gather: 2 lanes per edge-row, 16B vector copies (all offsets 16B-aligned)
    int row  = lane >> 1;
    int half = lane & 1;
    int e  = e0 + row;
    int iv = ev[e], ic = ec[e];
    // speculative prefetch of the next block's tile for this wave (global_prefetch_b8)
    __builtin_prefetch(ev + e + 64, 0, 1);
    __builtin_prefetch(ef16 + (size_t)(e + 64) * 8, 0, 1);
    const h8* pv = (const h8*)(vf16 + (size_t)iv * DV);
    const h8* pc = (const h8*)(cf16 + (size_t)ic * DV);
    const h8* pe = (const h8*)(ef16 + (size_t)e * 8);
    h8* dst = (h8*)(&A[row][0]);
    constexpr int NV   = DV / 8;                  // h8 chunks in vf/cf parts
    constexpr int PADS = (KPAD - 2 * DV - 8) / 8; // pad chunks (first carries the 1.0)
    if (half == 0) {
#pragma unroll
      for (int i = 0; i < NV; ++i) dst[i] = pv[i];
    } else {
#pragma unroll
      for (int i = 0; i < NV; ++i) dst[NV + i] = pc[i];
      dst[2 * NV] = pe[0];
      h8 z1 = {};
      z1[0] = (_Float16)1.f;                      // ones column at k = msg_in (bias row)
      dst[2 * NV + 1] = z1;
      h8 z = {};
#pragma unroll
      for (int i = 1; i < PADS; ++i) dst[2 * NV + 1 + i] = z;
    }
  }
  __builtin_amdgcn_wave_barrier();   // wave-private tile; LDS is in-order per wave

  int n  = lane & 15;
  int mb = (lane >> 4) * 8;

  // message A-fragments: shared by both sides, register-resident
  v16h af[KC1];
#pragma unroll
  for (int kc = 0; kc < KC1; ++kc) af[kc] = load_a_frag(&A[0][0], KPAD, kc * 32);

  const _Float16* W1s[2]  = {W1v, W1c};
  const _Float16* W2s[2]  = {W2v, W2c};
  const float*    b2s[2]  = {b2v, b2c};
  float*          aggs[2] = {aggv, aggc};
  const int*      idxs[2] = {ev, ec};

#pragma unroll
  for (int s = 0; s < 2; ++s) {
    const _Float16* W1 = W1s[s];
    const _Float16* W2 = W2s[s]; const float* b2 = b2s[s];

    // layer 1: KPAD -> 128 (bias folded in), relu, restage as f16 A-tile
    for (int nt = 0; nt < 8; ++nt) {
      v8f c = {};
#pragma unroll
      for (int kc = 0; kc < KC1; ++kc)
        c = wmma_f16(af[kc], load_b_packed(W1, kc * 8 + nt), c);
#pragma unroll
      for (int r = 0; r < 8; ++r)
        H[mb + r][nt * 16 + n] = (_Float16)fmaxf(c[r], 0.f);
    }
    __builtin_amdgcn_wave_barrier();

    // layer 2: 128 -> 64, atomic scatter-add (segment sum)
    v16h hf[4];
#pragma unroll
    for (int kc = 0; kc < 4; ++kc) hf[kc] = load_a_frag(&H[0][0], 128, kc * 32);

    const int* idx = idxs[s];
    float* agg = aggs[s];
    for (int nt = 0; nt < 4; ++nt) {
      v8f c = {};
#pragma unroll
      for (int kc = 0; kc < 4; ++kc)
        c = wmma_f16(hf[kc], load_b_packed(W2, kc * 4 + nt), c);
      float bias = b2[nt * 16 + n];
#pragma unroll
      for (int r = 0; r < 8; ++r) {
        int gi = idx[e0 + mb + r];
        atomicAdd(&agg[(size_t)gi * 64 + nt * 16 + n], c[r] + bias);
      }
    }
    __builtin_amdgcn_wave_barrier();
  }
}

// ---------------- fused node-update kernel (+ BN statistics) ----------------
template<int DV, int DIN>
__global__ __launch_bounds__(160)
void k_node_mlp(const float* __restrict__ feat,  // N x DV (f32)
                const float* __restrict__ agg,   // N x 64 (f32)
                const _Float16* __restrict__ W1, const float* __restrict__ b1,
                const _Float16* __restrict__ W2, const float* __restrict__ b2,
                float* __restrict__ out,         // N x 64 (pre-BN)
                float* __restrict__ sum, float* __restrict__ ssum) {
  constexpr int KC1 = DIN / 32;
  __shared__ _Float16 sA[5][16][DIN];
  __shared__ _Float16 sH[5][16][128];
  int wave = threadIdx.x >> 5;
  int lane = threadIdx.x & 31;
  int r0 = (blockIdx.x * 5 + wave) * 16;

  _Float16 (*A)[DIN] = sA[wave];
  _Float16 (*H)[128] = sH[wave];

  { // stage [feat | agg] as f16 (contiguous per-lane runs -> vectorized loads/stores)
    int row  = lane >> 1;
    int half = lane & 1;
    int gr = r0 + row;
    const float* pf = feat + (size_t)gr * DV;
    const float* pa = agg  + (size_t)gr * 64;
#pragma unroll
    for (int k = half * (DIN / 2); k < (half + 1) * (DIN / 2); ++k)
      A[row][k] = (_Float16)((k < DV) ? pf[k] : pa[k - DV]);
  }
  __builtin_amdgcn_wave_barrier();

  int n  = lane & 15;
  int mb = (lane >> 4) * 8;

  v16h af[KC1];
#pragma unroll
  for (int kc = 0; kc < KC1; ++kc) af[kc] = load_a_frag(&A[0][0], DIN, kc * 32);

  for (int nt = 0; nt < 8; ++nt) {
    v8f c = {};
#pragma unroll
    for (int kc = 0; kc < KC1; ++kc)
      c = wmma_f16(af[kc], load_b_packed(W1, kc * 8 + nt), c);
    float bias = b1[nt * 16 + n];
#pragma unroll
    for (int r = 0; r < 8; ++r)
      H[mb + r][nt * 16 + n] = (_Float16)fmaxf(c[r] + bias, 0.f);
  }
  __builtin_amdgcn_wave_barrier();

  v16h hf[4];
#pragma unroll
  for (int kc = 0; kc < 4; ++kc) hf[kc] = load_a_frag(&H[0][0], 128, kc * 32);

  for (int nt = 0; nt < 4; ++nt) {
    v8f c = {};
#pragma unroll
    for (int kc = 0; kc < 4; ++kc)
      c = wmma_f16(hf[kc], load_b_packed(W2, kc * 4 + nt), c);
    int col = nt * 16 + n;
    float bias = b2[col];
    float s = 0.f, ss = 0.f;
#pragma unroll
    for (int r = 0; r < 8; ++r) {
      float y = c[r] + bias;
      out[(size_t)(r0 + mb + r) * 64 + col] = y;
      s += y; ss += y * y;
    }
    atomicAdd(&sum[col], s);
    atomicAdd(&ssum[col], ss);
  }
}

// ---------------- host orchestration ----------------

extern "C" void kernel_launch(void* const* d_in, const int* in_sizes, int n_in,
                              void* d_out, int out_size, void* d_ws, size_t ws_size,
                              hipStream_t stream) {
  const float* var_f  = (const float*)d_in[0];   // 50000 x 32
  const float* cons_f = (const float*)d_in[1];   // 50000 x 32
  const float* edge_f = (const float*)d_in[2];   // 800000 x 8
  // params: per iter, sorted-dict leaf order: cbn(g,b) cm(W1,b1,W2,b2) cu(...) vbn(g,b) vm(...) vu(...)
  const float* P[40];
  for (int i = 0; i < 40; ++i) P[i] = (const float*)d_in[3 + i];
  const int* edge_var  = (const int*)d_in[43];
  const int* edge_cons = (const int*)d_in[44];

  // scratch carving (all sizes 256B multiples)
  char* w = (char*)d_ws;
  size_t off = 0;
  auto carve = [&](size_t bytes) {
    void* p = w + off;
    off = (off + bytes + 255) & ~(size_t)255;
    return p;
  };
  float*     vf_cur = (float*)carve((size_t)N_VAR * 64 * 4);
  float*     cf_cur = (float*)carve((size_t)N_CONS * 64 * 4);
  _Float16*  vf16   = (_Float16*)carve((size_t)N_VAR * 64 * 2);
  _Float16*  cf16   = (_Float16*)carve((size_t)N_CONS * 64 * 2);
  _Float16*  ef16   = (_Float16*)carve((size_t)N_EDGE * 8 * 2);
  float*     agg_v  = (float*)carve((size_t)N_VAR * 64 * 4);   // agg_v, agg_c, stats contiguous
  float*     agg_c  = (float*)carve((size_t)N_CONS * 64 * 4);
  float*     stats  = (float*)carve(4 * 64 * 4);               // sum_v | ss_v | sum_c | ss_c
  float*     new_v  = (float*)carve((size_t)N_VAR * 64 * 4);
  float*     new_c  = (float*)carve((size_t)N_CONS * 64 * 4);
  _Float16*  w1vm = (_Float16*)carve(160 * 128 * 2);
  _Float16*  w1cm = (_Float16*)carve(160 * 128 * 2);
  _Float16*  w2vm = (_Float16*)carve(128 * 64 * 2);
  _Float16*  w2cm = (_Float16*)carve(128 * 64 * 2);
  _Float16*  w1vu = (_Float16*)carve(128 * 128 * 2);
  _Float16*  w1cu = (_Float16*)carve(128 * 128 * 2);
  _Float16*  w2vu = (_Float16*)carve(128 * 64 * 2);
  _Float16*  w2cu = (_Float16*)carve(128 * 64 * 2);

  auto g1 = [](int n) { return (n + 255) / 256; };

  // one-time f16 staging of iteration-0 inputs
  k_f32_to_f16<<<g1(N_VAR * 32), 256, 0, stream>>>(vf16, var_f, N_VAR * 32);
  k_f32_to_f16<<<g1(N_CONS * 32), 256, 0, stream>>>(cf16, cons_f, N_CONS * 32);
  k_f32_to_f16<<<g1(N_EDGE * 8), 256, 0, stream>>>(ef16, edge_f, N_EDGE * 8);

  for (int it = 0; it < 2; ++it) {
    const int dv = (it == 0) ? 32 : 64;
    const int msg_in = 2 * dv + 8;                 // 72 / 136
    const int kpad1  = (it == 0) ? 96 : 160;       // K padded to mult of 32 (slack holds bias row)
    const int din    = dv + 64;                    // 96 / 128

    const float* cbn_g = P[it * 20 + 0];  const float* cbn_b = P[it * 20 + 1];
    const float* cmW1  = P[it * 20 + 2];  const float* cmb1  = P[it * 20 + 3];
    const float* cmW2  = P[it * 20 + 4];  const float* cmb2  = P[it * 20 + 5];
    const float* cuW1  = P[it * 20 + 6];  const float* cub1  = P[it * 20 + 7];
    const float* cuW2  = P[it * 20 + 8];  const float* cub2  = P[it * 20 + 9];
    const float* vbn_g = P[it * 20 + 10]; const float* vbn_b = P[it * 20 + 11];
    const float* vmW1  = P[it * 20 + 12]; const float* vmb1  = P[it * 20 + 13];
    const float* vmW2  = P[it * 20 + 14]; const float* vmb2  = P[it * 20 + 15];
    const float* vuW1  = P[it * 20 + 16]; const float* vub1  = P[it * 20 + 17];
    const float* vuW2  = P[it * 20 + 18]; const float* vub2  = P[it * 20 + 19];

    // weight prep: f16, K-padded, WMMA-fragment-major; cons messenger W1 row-swapped;
    // messenger layer-1 biases folded in as weight row msg_in (pairs with A's 1.0 column)
    k_pack_w<<<g1(kpad1 * 128), 256, 0, stream>>>(w1vm, vmW1, vmb1, msg_in, kpad1, 128, 0);
    k_pack_w<<<g1(kpad1 * 128), 256, 0, stream>>>(w1cm, cmW1, cmb1, msg_in, kpad1, 128, dv);
    k_pack_w<<<g1(128 * 64), 256, 0, stream>>>(w2vm, vmW2, nullptr, 128, 128, 64, 0);
    k_pack_w<<<g1(128 * 64), 256, 0, stream>>>(w2cm, cmW2, nullptr, 128, 128, 64, 0);
    k_pack_w<<<g1(din * 128), 256, 0, stream>>>(w1vu, vuW1, nullptr, din, din, 128, 0);
    k_pack_w<<<g1(din * 128), 256, 0, stream>>>(w1cu, cuW1, nullptr, din, din, 128, 0);
    k_pack_w<<<g1(128 * 64), 256, 0, stream>>>(w2vu, vuW2, nullptr, 128, 128, 64, 0);
    k_pack_w<<<g1(128 * 64), 256, 0, stream>>>(w2cu, cuW2, nullptr, 128, 128, 64, 0);

    // zero agg_v + agg_c + stats (contiguous)
    k_zero_f32<<<g1(N_VAR * 64 + N_CONS * 64 + 256), 256, 0, stream>>>(
        agg_v, N_VAR * 64 + N_CONS * 64 + 256);

    // fused edge messengers + segment-sum: 800000 edges = 12500 blocks * 4 waves * 16
    if (it == 0)
      k_edge_mlp<32, 96><<<N_EDGE / 64, 128, 0, stream>>>(
          vf16, cf16, ef16, edge_var, edge_cons,
          w1vm, w2vm, vmb2, w1cm, w2cm, cmb2, agg_v, agg_c);
    else
      k_edge_mlp<64, 160><<<N_EDGE / 64, 128, 0, stream>>>(
          vf16, cf16, ef16, edge_var, edge_cons,
          w1vm, w2vm, vmb2, w1cm, w2cm, cmb2, agg_v, agg_c);

    // node updates (+ BN stats): 50000 nodes = 625 blocks * 5 waves * 16
    const float* vfeat = (it == 0) ? var_f  : vf_cur;
    const float* cfeat = (it == 0) ? cons_f : cf_cur;
    if (it == 0) {
      k_node_mlp<32, 96><<<N_VAR / 80, 160, 0, stream>>>(
          vfeat, agg_v, w1vu, vub1, w2vu, vub2, new_v, stats + 0, stats + 64);
      k_node_mlp<32, 96><<<N_CONS / 80, 160, 0, stream>>>(
          cfeat, agg_c, w1cu, cub1, w2cu, cub2, new_c, stats + 128, stats + 192);
    } else {
      k_node_mlp<64, 128><<<N_VAR / 80, 160, 0, stream>>>(
          vfeat, agg_v, w1vu, vub1, w2vu, vub2, new_v, stats + 0, stats + 64);
      k_node_mlp<64, 128><<<N_CONS / 80, 160, 0, stream>>>(
          cfeat, agg_c, w1cu, cub1, w2cu, cub2, new_c, stats + 128, stats + 192);
    }

    // BatchNorm apply; last iteration streams straight into d_out (vf then cf)
    bool last = (it == 1);
    float* vdst = last ? (float*)d_out : vf_cur;
    float* cdst = last ? ((float*)d_out + (size_t)N_VAR * 64) : cf_cur;
    k_bn_apply<<<g1(N_VAR * 64), 256, 0, stream>>>(
        new_v, stats + 0, stats + 64, vbn_g, vbn_b,
        vdst, last ? (_Float16*)nullptr : vf16, N_VAR * 64, 1.0f / N_VAR);
    k_bn_apply<<<g1(N_CONS * 64), 256, 0, stream>>>(
        new_c, stats + 128, stats + 192, cbn_g, cbn_b,
        cdst, last ? (_Float16*)nullptr : cf16, N_CONS * 64, 1.0f / N_CONS);
  }
  (void)in_sizes; (void)n_in; (void)out_size; (void)ws_size;
}
